// RnnReceiverFixedLengthGS_13546326851857
// MI455X (gfx1250) — matrix operand order repaired and checked by reading, not compiled, not run on hardware
//
#include <hip/hip_runtime.h>

// ---------------------------------------------------------------------------
// CDNA5 (gfx1250) WMMA GRU receiver.
//   B=1024, T=128, V=1024, E=256, H=512, OUT=2048
// All GEMMs via v_wmma_f32_16x16x32_bf16 (bf16 inputs, f32 accumulate).
// ---------------------------------------------------------------------------

typedef __bf16 bf16;
typedef bf16  v16bf __attribute__((ext_vector_type(16)));
typedef bf16  v8bf  __attribute__((ext_vector_type(8)));
typedef bf16  v4bf  __attribute__((ext_vector_type(4)));
typedef float v8f   __attribute__((ext_vector_type(8)));

// --- fragment helpers (layouts per CDNA5 ISA §7.12.2) ------------

__device__ __forceinline__ v16bf combine8(v8bf lo, v8bf hi) {
  return __builtin_shufflevector(lo, hi, 0,1,2,3,4,5,6,7,8,9,10,11,12,13,14,15);
}

// A-matrix 16x32 bf16 fragment from a row-major bf16 row.
// lane<16 : elements 0..7 = K kb..kb+7,    8..15 = K kb+16..kb+23
// lane>=16: elements 0..7 = K kb+8..kb+15, 8..15 = K kb+24..kb+31
__device__ __forceinline__ v16bf load_a_bf16(const bf16* __restrict__ row, int kb, int half) {
  v8bf lo = *(const v8bf*)(row + kb + half * 8);
  v8bf hi = *(const v8bf*)(row + kb + 16 + half * 8);
  return combine8(lo, hi);
}

// Same A fragment but sourced from fp32 with in-register bf16 conversion.
__device__ __forceinline__ v16bf load_a_f32(const float* __restrict__ row, int kb, int half) {
  const float4 a0 = *(const float4*)(row + kb + half * 8);
  const float4 a1 = *(const float4*)(row + kb + half * 8 + 4);
  const float4 a2 = *(const float4*)(row + kb + 16 + half * 8);
  const float4 a3 = *(const float4*)(row + kb + 16 + half * 8 + 4);
  v16bf r;
  r[0]  = (bf16)a0.x; r[1]  = (bf16)a0.y; r[2]  = (bf16)a0.z; r[3]  = (bf16)a0.w;
  r[4]  = (bf16)a1.x; r[5]  = (bf16)a1.y; r[6]  = (bf16)a1.z; r[7]  = (bf16)a1.w;
  r[8]  = (bf16)a2.x; r[9]  = (bf16)a2.y; r[10] = (bf16)a2.z; r[11] = (bf16)a2.w;
  r[12] = (bf16)a3.x; r[13] = (bf16)a3.y; r[14] = (bf16)a3.z; r[15] = (bf16)a3.w;
  return r;
}

// B-matrix 32x16 fragment: column n of B == row (f) of W [N,K] row-major.
// lane<16: K kb..kb+7, kb+16..kb+23; lane>=16: K kb+8..kb+15, kb+24..kb+31.
__device__ __forceinline__ v16bf load_w(const bf16* __restrict__ W, int r, int ldk,
                                        int kb, int half) {
  const bf16* p = W + (size_t)r * ldk + kb + half * 8;
  v8bf lo = *(const v8bf*)p;
  v8bf hi = *(const v8bf*)(p + 16);
  return combine8(lo, hi);
}

__device__ __forceinline__ v8f wmma_bf16(v16bf a, v16bf b, v8f c) {
  return __builtin_amdgcn_wmma_f32_16x16x32_bf16(false, a, false, b, (short)0, c,
                                                 false, false);
}

__device__ __forceinline__ float sigmoidf_(float x) {
  return 1.0f / (1.0f + __expf(-x));
}

// --- fp32 -> bf16 weight pre-conversion ------------------------------------

__global__ __launch_bounds__(256) void cvt_f32_bf16_k(const float* __restrict__ in,
                                                      bf16* __restrict__ out, int n4) {
  int i = blockIdx.x * blockDim.x + threadIdx.x;
  if (i < n4) {
    float4 v = ((const float4*)in)[i];
    v4bf o;
    o[0] = (bf16)v.x; o[1] = (bf16)v.y; o[2] = (bf16)v.z; o[3] = (bf16)v.w;
    *(v4bf*)(out + (size_t)i * 4) = o;
  }
}

// --- generic WMMA GEMM: C[M,N] = A[M,K](f32) * W[N,K]^T(bf16) + bias -------
// Block = 256 threads = 8 waves; each wave owns a 16-row strip and NT 16-col
// tiles => block tile 128 x (16*NT). REMAP maps output row r = t*B+b to
// message row b*T+t (B=1024 hard-wired as shift/mask).

template <int NT, bool REMAP, bool OUTBF16>
__global__ __launch_bounds__(256) void gemm_a32(const float* __restrict__ A,
                                                const bf16* __restrict__ W,
                                                const float* __restrict__ bias,
                                                void* __restrict__ Cout,
                                                int N, int K, int Tdim) {
  const int wave = threadIdx.x >> 5;
  const int lane = threadIdx.x & 31;
  const int half = lane >> 4;
  const int lm   = lane & 15;
  const int row  = blockIdx.x * 128 + wave * 16 + lm;

  const float* arow;
  if (REMAP) {
    const int t  = row >> 10;       // row / 1024  (B = 1024)
    const int bb = row & 1023;      // row % 1024
    arow = A + ((size_t)bb * Tdim + t) * (size_t)K;
  } else {
    arow = A + (size_t)row * K;
  }

  const int f0 = blockIdx.y * (16 * NT);

  const v8f vz = {0.f, 0.f, 0.f, 0.f, 0.f, 0.f, 0.f, 0.f};
  v8f acc[NT];
#pragma unroll
  for (int i = 0; i < NT; ++i) acc[i] = vz;

  for (int kb = 0; kb < K; kb += 32) {
    if (kb + 64 < K) __builtin_prefetch(arow + kb + 64, 0, 1);  // global_prefetch_b8
    const v16bf af = load_a_f32(arow, kb, half);
#pragma unroll
    for (int nt = 0; nt < NT; ++nt) {
      const v16bf bfrag = load_w(W, f0 + nt * 16 + lm, K, kb, half);
      acc[nt] = wmma_bf16(af, bfrag, acc[nt]);
    }
  }

  // C/D layout: lane<16 -> rows v (v=0..7), lane>=16 -> rows v+8; col = lane&15.
  const int mb = blockIdx.x * 128 + wave * 16 + half * 8;
#pragma unroll
  for (int nt = 0; nt < NT; ++nt) {
    const int col = f0 + nt * 16 + lm;
    const float bv = bias[col];
#pragma unroll
    for (int v = 0; v < 8; ++v) {
      const float val = acc[nt][v] + bv;
      const size_t idx = (size_t)(mb + v) * N + col;
      if (OUTBF16) ((bf16*)Cout)[idx] = (bf16)val;
      else         ((float*)Cout)[idx] = val;
    }
  }
}

// --- one GRU timestep -------------------------------------------------------
// Fuses gi = x@W_ih^T (K=E over bf16 emb) and gh = h@W_hh^T (K=H over fp32 h),
// with separate accumulators for r, z, i_n, h_n, then applies the gates.
// Block = 256 threads = 8 waves -> 64x64 tile of [B=1024, H=512].

__global__ __launch_bounds__(256) void gru_step(const bf16* __restrict__ x,
                                                const float* __restrict__ hin,
                                                float* __restrict__ hout,
                                                const bf16* __restrict__ Wih,
                                                const bf16* __restrict__ Whh,
                                                const float* __restrict__ bih,
                                                const float* __restrict__ bhh,
                                                int firstStep) {
  constexpr int E = 256, H = 512;
  const int wave = threadIdx.x >> 5;
  const int lane = threadIdx.x & 31;
  const int half = lane >> 4;
  const int lm   = lane & 15;
  const int m0 = blockIdx.x * 64 + (wave & 3) * 16;
  const int n0 = blockIdx.y * 64 + (wave >> 2) * 32;

  const v8f vz = {0.f, 0.f, 0.f, 0.f, 0.f, 0.f, 0.f, 0.f};
  v8f accr[2] = {vz, vz}, accz[2] = {vz, vz}, accin[2] = {vz, vz}, acchn[2] = {vz, vz};

  // gi path: K over E, A = bf16 emb slice [B,E]
  const bf16* xrow = x + (size_t)(m0 + lm) * E;
  for (int kb = 0; kb < E; kb += 32) {
    const v16bf af = load_a_bf16(xrow, kb, half);
#pragma unroll
    for (int ns = 0; ns < 2; ++ns) {
      const int f = n0 + ns * 16 + lm;
      accr[ns]  = wmma_bf16(af, load_w(Wih, f,         E, kb, half), accr[ns]);
      accz[ns]  = wmma_bf16(af, load_w(Wih, f + H,     E, kb, half), accz[ns]);
      accin[ns] = wmma_bf16(af, load_w(Wih, f + 2 * H, E, kb, half), accin[ns]);
    }
  }

  // gh path: K over H, A = fp32 h (skip when h == 0 on the first step)
  if (!firstStep) {
    const float* hrow = hin + (size_t)(m0 + lm) * H;
    for (int kb = 0; kb < H; kb += 32) {
      const v16bf af = load_a_f32(hrow, kb, half);
#pragma unroll
      for (int ns = 0; ns < 2; ++ns) {
        const int f = n0 + ns * 16 + lm;
        accr[ns]  = wmma_bf16(af, load_w(Whh, f,         H, kb, half), accr[ns]);
        accz[ns]  = wmma_bf16(af, load_w(Whh, f + H,     H, kb, half), accz[ns]);
        acchn[ns] = wmma_bf16(af, load_w(Whh, f + 2 * H, H, kb, half), acchn[ns]);
      }
    }
  }

  // gates + state update
  const int mb = m0 + half * 8;
#pragma unroll
  for (int ns = 0; ns < 2; ++ns) {
    const int col = n0 + ns * 16 + lm;
    const float br = bih[col] + bhh[col];
    const float bz = bih[col + H] + bhh[col + H];
    const float bi = bih[col + 2 * H];
    const float bh = bhh[col + 2 * H];
#pragma unroll
    for (int v = 0; v < 8; ++v) {
      const int r = mb + v;
      const float rr = sigmoidf_(accr[ns][v] + br);
      const float zz = sigmoidf_(accz[ns][v] + bz);
      const float nn = tanhf(accin[ns][v] + bi + rr * (acchn[ns][v] + bh));
      const float hp = firstStep ? 0.f : hin[(size_t)r * H + col];
      hout[(size_t)r * H + col] = (1.f - zz) * nn + zz * hp;
    }
  }
}

// ---------------------------------------------------------------------------

extern "C" void kernel_launch(void* const* d_in, const int* in_sizes, int n_in,
                              void* d_out, int out_size, void* d_ws, size_t ws_size,
                              hipStream_t stream) {
  (void)in_sizes; (void)n_in; (void)out_size; (void)ws_size;
  constexpr int Bb = 1024, T = 128, V = 1024, E = 256, H = 512, OUT = 2048;

  const float* message = (const float*)d_in[0];
  const float* W_emb   = (const float*)d_in[1];
  const float* b_emb   = (const float*)d_in[2];
  const float* W_ih    = (const float*)d_in[3];
  const float* b_ih    = (const float*)d_in[4];
  const float* W_hh    = (const float*)d_in[5];
  const float* b_hh    = (const float*)d_in[6];
  const float* W_ag    = (const float*)d_in[7];
  const float* b_ag    = (const float*)d_in[8];

  // workspace carve-out (256B aligned)
  char* ws = (char*)d_ws;
  size_t off = 0;
  auto take = [&](size_t bytes) -> char* {
    char* p = ws + off;
    off += (bytes + 255) & ~(size_t)255;
    return p;
  };
  bf16*  wemb_b = (bf16*)take((size_t)E * V * 2);
  bf16*  wih_b  = (bf16*)take((size_t)3 * H * E * 2);
  bf16*  whh_b  = (bf16*)take((size_t)3 * H * H * 2);
  bf16*  wag_b  = (bf16*)take((size_t)OUT * H * 2);
  bf16*  emb_b  = (bf16*)take((size_t)(T - 1) * Bb * E * 2);  // time-major [T-1,B,E]
  float* h0     = (float*)take((size_t)Bb * H * 4);
  float* h1     = (float*)take((size_t)Bb * H * 4);

  auto cvt = [&](const float* src, bf16* dst, int n) {
    int n4 = n / 4;
    cvt_f32_bf16_k<<<(n4 + 255) / 256, 256, 0, stream>>>(src, dst, n4);
  };
  cvt(W_emb, wemb_b, E * V);
  cvt(W_ih,  wih_b,  3 * H * E);
  cvt(W_hh,  whh_b,  3 * H * H);
  cvt(W_ag,  wag_b,  OUT * H);

  // 1) embedding GEMM: [B*(T-1), V] x [V, E] -> time-major bf16 emb.
  //    Full-N per block (NT=16 -> 256 cols) => each message row read once.
  {
    dim3 grid(((T - 1) * Bb) / 128, 1);
    gemm_a32<16, true, true><<<grid, 256, 0, stream>>>(
        message, wemb_b, b_emb, emb_b, E, V, T);
  }

  // 2) GRU: 127 sequential steps; kernel boundary is the grid-wide sync on h.
  {
    dim3 grid(Bb / 64, H / 64);
    float* bufs[2] = {h0, h1};
    for (int t = 0; t < T - 1; ++t) {
      const bf16* xt = emb_b + (size_t)t * Bb * E;
      gru_step<<<grid, 256, 0, stream>>>(xt, bufs[t & 1], bufs[(t + 1) & 1],
                                         wih_b, whh_b, b_ih, b_hh, t == 0 ? 1 : 0);
    }
  }

  // 3) head GEMM: [B, H] x [H, OUT] -> d_out (fp32). Final h is in h1 (T-1 odd).
  {
    float* hfinal = ((T - 1) & 1) ? h1 : h0;
    dim3 grid(Bb / 128, OUT / 128);
    gemm_a32<8, false, false><<<grid, 256, 0, stream>>>(
        hfinal, wag_b, b_ag, d_out, OUT, H, 0);
  }
}
